// NetVLAD_40544491274592
// MI455X (gfx1250) — matrix operand order repaired and checked by reading, not compiled, not run on hardware
//
#include <hip/hip_runtime.h>
#include <hip/hip_bf16.h>
#include <stdint.h>

// Problem constants (from reference setup_inputs)
#define N_   4
#define C_   128
#define H_   40
#define W_   40
#define K_   64
#define P_   (H_*W_)      // 1600 spatial positions
#define HO_  18
#define WO_  18
#define WIN_ (HO_*WO_)    // 324 windows
#define EPS_ 1e-12f

typedef __attribute__((ext_vector_type(16))) __bf16 v16bf;
typedef __attribute__((ext_vector_type(8)))  float v8f;

__device__ __forceinline__ unsigned short f2bf(float f) {
  union { float f; unsigned u; } v; v.f = f;
  unsigned u = v.u;
  unsigned r = u + 0x7FFFu + ((u >> 16) & 1u);   // round-to-nearest-even
  return (unsigned short)(r >> 16);
}
__device__ __forceinline__ float bf2f(unsigned short h) {
  union { unsigned u; float f; } v; v.u = ((unsigned)h) << 16; return v.f;
}

// Generic pointers to LDS carry the LDS byte offset in their low 32 bits.
__device__ __forceinline__ unsigned lds_off(const void* p) {
  return (unsigned)(unsigned long long)p;
}
// CDNA5 async block copy global -> LDS (ASYNCcnt-tracked, no VGPR staging).
__device__ __forceinline__ void async_b128(unsigned lds_byte_off, const void* gptr) {
  asm volatile("global_load_async_to_lds_b128 %0, %1, off"
               :: "v"(lds_byte_off), "v"((unsigned long long)gptr)
               : "memory");
}
__device__ __forceinline__ void wait_async() {
  asm volatile("s_wait_asynccnt 0" ::: "memory");
}

union bfrag { uint4 q[2]; v16bf v; };

// ---- WMMA fragment loaders per CDNA5 ISA 7.12.2 (wave32), 2x ds_load_b128 each ----
// A 16x32 bf16 (MxK), tile K-contiguous (row stride `stride` elems).
// Lane l: row m=l&15; VGPRs 0-3 hold K=khi..khi+7, VGPRs 4-7 hold K=16+khi..16+khi+7 (khi = (l>=16)?8:0)
__device__ __forceinline__ v16bf load_a_bf16(const unsigned short* s, int stride, int lane) {
  int m   = lane & 15;
  int khi = (lane >> 4) << 3;
  bfrag f;
  f.q[0] = *(const uint4*)(s + m * stride + khi);
  f.q[1] = *(const uint4*)(s + m * stride + 16 + khi);
  return f.v;
}
// B 32x16 bf16 (KxN), tile stored TRANSPOSED (element (k,n) at s[n*stride + k], K-contiguous).
// Lane l: col n=l&15; VGPRs 0-3 hold K=khi..khi+7, VGPRs 4-7 hold K=16+khi..16+khi+7 (khi = (l>=16)?8:0)
__device__ __forceinline__ v16bf load_bt_bf16(const unsigned short* s, int stride, int lane) {
  int nn   = lane & 15;
  int khi16 = (lane >> 4) << 3;
  bfrag f;
  f.q[0] = *(const uint4*)(s + nn * stride + khi16);
  f.q[1] = *(const uint4*)(s + nn * stride + khi16 + 16);
  return f.v;
}
// C/D 16x16 f32: lane l holds col N=l&15; VGPR v: row M = v + (l>=16?8:0)
__device__ __forceinline__ void store_c_f32(float* s, int stride, int lane, v8f c) {
  int n  = lane & 15;
  int mo = (lane >> 4) << 3;
#pragma unroll
  for (int v = 0; v < 8; ++v) s[(v + mo) * stride + n] = c[v];
}
__device__ __forceinline__ v8f wmma_bf16(v16bf a, v16bf b, v8f c) {
  return __builtin_amdgcn_wmma_f32_16x16x32_bf16(false, a, false, b, (short)0, c, false, false);
}

// ---------------- Kernel 0: conv_w -> bf16 (identity [k][c] layout == transposed-B tile), zero S ----------------
__global__ void k_prep(const float* __restrict__ conv_w, unsigned short* __restrict__ wb,
                       float* __restrict__ S) {
  int idx = blockIdx.x * blockDim.x + threadIdx.x;   // 8192 = K_*C_
  if (idx < K_ * C_) wb[idx] = f2bf(conv_w[idx]);
  if (idx < N_ * K_) S[idx] = 0.f;
}

// ---------------- Kernel 1: descriptor L2-normalize; write x̂ bf16 in BOTH layouts ----------------
// xn_pm[n][p][c]  (c-contiguous: A-tiles for the assignment GEMM)
// xn_cm[n][c][p]  (p-contiguous: transposed B-tiles for the VLAD GEMMs, async-copyable)
__global__ void k_normalize(const float* __restrict__ x,
                            unsigned short* __restrict__ xn_pm,
                            unsigned short* __restrict__ xn_cm) {
  int pg = blockIdx.x * blockDim.x + threadIdx.x;    // 6400 positions, exact grid
  int n = pg / P_, p = pg % P_;
  const float* xp = x + (size_t)n * C_ * P_ + p;
  float ss = 0.f;
#pragma unroll 8
  for (int c = 0; c < C_; ++c) { float v = xp[c * P_]; ss += v * v; }
  float rn = 1.f / fmaxf(sqrtf(ss), EPS_);
  unsigned short* o = xn_pm + ((size_t)n * P_ + p) * C_;
#pragma unroll 8
  for (int c = 0; c < C_; ++c) {
    unsigned short h = f2bf(xp[c * P_] * rn);
    o[c] = h;
    xn_cm[((size_t)n * C_ + c) * P_ + p] = h;
  }
}

// ---------------- Kernel 2: soft-assign GEMM (WMMA, async-staged) + softmax ----------------
__global__ void k_assign(const unsigned short* __restrict__ xn_pm, const unsigned short* __restrict__ wb,
                         unsigned short* __restrict__ sa, float* __restrict__ S) {
  __shared__ alignas(16) unsigned short Asx[64 * C_];   // 64 pos x 128 ch (16 KB)
  __shared__ alignas(16) unsigned short Ws[K_ * C_];    // 64 k  x 128 ch (16 KB) == transposed B
  __shared__ float Ls[64 * K_];                         // logits -> probs (16 KB)
  int tid = threadIdx.x, lane = tid & 31, w = tid >> 5;
  int n = blockIdx.x / (P_ / 64), p0 = (blockIdx.x % (P_ / 64)) * 64;

  const unsigned short* src = xn_pm + ((size_t)n * P_ + p0) * C_;   // 16 KB contiguous
#pragma unroll
  for (int i = 0; i < 8; ++i) {            // 1024 x b128: x̂ tile
    int q = tid + i * 128;
    async_b128(lds_off(&Asx[q * 8]), src + q * 8);
  }
#pragma unroll
  for (int i = 0; i < 8; ++i) {            // 1024 x b128: weights
    int q = tid + i * 128;
    async_b128(lds_off(&Ws[q * 8]), wb + q * 8);
  }
  wait_async();
  __syncthreads();

  v8f acc[4];
#pragma unroll
  for (int nt = 0; nt < 4; ++nt) acc[nt] = v8f{0,0,0,0,0,0,0,0};
#pragma unroll
  for (int kk = 0; kk < 4; ++kk) {
    v16bf af = load_a_bf16(Asx + (16 * w) * C_ + 32 * kk, C_, lane);
#pragma unroll
    for (int nt = 0; nt < 4; ++nt) {
      v16bf bf = load_bt_bf16(Ws + (16 * nt) * C_ + 32 * kk, C_, lane);
      acc[nt] = wmma_bf16(af, bf, acc[nt]);
    }
  }
#pragma unroll
  for (int nt = 0; nt < 4; ++nt) store_c_f32(Ls + (16 * w) * K_ + 16 * nt, K_, lane, acc[nt]);
  __syncthreads();

  if (tid < 64) {                           // softmax over K for row (position) tid
    float* row = Ls + tid * K_;
    float m = row[0];
#pragma unroll 8
    for (int k = 1; k < K_; ++k) m = fmaxf(m, row[k]);
    float s = 0.f;
#pragma unroll 8
    for (int k = 0; k < K_; ++k) { float e = __expf(row[k] - m); row[k] = e; s += e; }
    float is = 1.f / s;
#pragma unroll 8
    for (int k = 0; k < K_; ++k) {
      float pr = row[k] * is; row[k] = pr;
      sa[((size_t)n * K_ + k) * P_ + p0 + tid] = f2bf(pr);   // cluster-major (p-contiguous rows)
    }
  }
  __syncthreads();
  if (tid < K_) {                           // partial Σ_p a[k,p]
    float s = 0.f;
#pragma unroll 8
    for (int r = 0; r < 64; ++r) s += Ls[r * K_ + tid];
    atomicAdd(&S[n * K_ + tid], s);
  }
}

// ---------------- Kernel 3: global VLAD = SA·X̂ᵀ, async double-buffered pipeline ----------------
__global__ void k_vlad_full(const unsigned short* __restrict__ sa, const unsigned short* __restrict__ xn_cm,
                            const float* __restrict__ cent, const float* __restrict__ S,
                            float* __restrict__ out2) {
  __shared__ alignas(16) unsigned short sas[2][K_ * 32];   // 64x32 A tiles   (2 x 4 KB)
  __shared__ alignas(16) unsigned short xct[2][C_ * 32];   // 128x32 B tiles (transposed, 2 x 8 KB)
  __shared__ float vf[K_ * C_];                            // 64x128 result (32 KB)
  __shared__ float rr[K_], red2[K_];
  __shared__ float gval;
  int tid = threadIdx.x, lane = tid & 31, w = tid >> 5;
  int n = blockIdx.x;

  auto issue = [&](int kk, int buf) {
#pragma unroll
    for (int i = 0; i < 2; ++i) {          // SA tile: 256 x b128
      int q = tid + i * 128, k = q >> 2, part = q & 3;
      async_b128(lds_off(&sas[buf][k * 32 + part * 8]),
                 sa + ((size_t)n * K_ + k) * P_ + kk * 32 + part * 8);
    }
#pragma unroll
    for (int i = 0; i < 4; ++i) {          // X̂ tile (channel-major): 512 x b128
      int q = tid + i * 128, c = q >> 2, part = q & 3;
      async_b128(lds_off(&xct[buf][c * 32 + part * 8]),
                 xn_cm + ((size_t)n * C_ + c) * P_ + kk * 32 + part * 8);
    }
  };

  v8f acc[8];
#pragma unroll
  for (int nt = 0; nt < 8; ++nt) acc[nt] = v8f{0,0,0,0,0,0,0,0};

  issue(0, 0);
  for (int kk = 0; kk < P_ / 32; ++kk) {    // 50 K-steps, DMA(kk+1) overlaps WMMA(kk)
    wait_async();
    __syncthreads();
    if (kk + 1 < P_ / 32) issue(kk + 1, (kk + 1) & 1);
    int buf = kk & 1;
    v16bf af = load_a_bf16(&sas[buf][(16 * w) * 32], 32, lane);
#pragma unroll
    for (int nt = 0; nt < 8; ++nt) {
      v16bf bf = load_bt_bf16(&xct[buf][(16 * nt) * 32], 32, lane);
      acc[nt] = wmma_bf16(af, bf, acc[nt]);
    }
  }
#pragma unroll
  for (int nt = 0; nt < 8; ++nt) store_c_f32(vf + (16 * w) * C_ + 16 * nt, C_, lane, acc[nt]);
  __syncthreads();

#pragma unroll 4
  for (int k = 0; k < K_; ++k) {            // residual correction: − c[k,c]·S[n,k]
    float v = vf[k * C_ + tid] - cent[k * C_ + tid] * S[n * K_ + k];
    vf[k * C_ + tid] = v;
  }
  __syncthreads();
  if (tid < K_) {                           // intra-norm over C
    float ss = 0.f;
#pragma unroll 8
    for (int c = 0; c < C_; ++c) { float v = vf[tid * C_ + c]; ss += v * v; }
    float r = 1.f / fmaxf(sqrtf(ss), EPS_);
    rr[tid] = r; red2[tid] = ss * r * r;
  }
  __syncthreads();
  if (tid == 0) {
    float g2 = 0.f;
    for (int k = 0; k < K_; ++k) g2 += red2[k];
    gval = 1.f / fmaxf(sqrtf(g2), EPS_);
  }
  __syncthreads();
  float g = gval;
#pragma unroll 4
  for (int k = 0; k < K_; ++k)
    out2[(size_t)n * (K_ * C_) + k * C_ + tid] = vf[k * C_ + tid] * rr[k] * g;
}

// ---------------- Kernel 4: regional VLAD, one 5x5 window per block (WMMA 64x128x32) ----------------
__global__ void k_windows(const unsigned short* __restrict__ sa, const unsigned short* __restrict__ xn_cm,
                          const float* __restrict__ cent, float* __restrict__ out) {
  __shared__ alignas(16) unsigned short sas[K_ * 32];   // SA_win 64x(25→32)           (4 KB)
  __shared__ alignas(16) unsigned short xct[C_ * 32];   // X̂_win transposed 128x(25→32) (8 KB)
  __shared__ float Ts[K_];
  __shared__ float ps[K_ * C_];                          // 32 KB
  __shared__ float rr[K_], red2[K_];
  __shared__ float gval;
  int tid = threadIdx.x, lane = tid & 31, w = tid >> 5;
  int b = blockIdx.x, n = b / WIN_, win = b % WIN_;
  int h0 = (win / WO_) * 2, w0 = (win % WO_) * 2;

  __builtin_prefetch(cent + tid, 0, 3);     // global_prefetch for the fp32 centroid table

#pragma unroll
  for (int i = 0; i < 16; ++i) {            // stage SA_win rows (pad j>=25 with 0)
    int idx = tid + i * 128, k = idx >> 5, j = idx & 31;
    unsigned short v = 0;
    if (j < 25) {
      int p = (h0 + j / 5) * W_ + (w0 + j % 5);
      v = sa[((size_t)n * K_ + k) * P_ + p];
    }
    sas[idx] = v;
  }
#pragma unroll
  for (int j = 0; j < 32; ++j) {            // stage X̂_win channel-major (thread = channel)
    unsigned short v = 0;
    if (j < 25) {
      int p = (h0 + j / 5) * W_ + (w0 + j % 5);
      v = xn_cm[((size_t)n * C_ + tid) * P_ + p];
    }
    xct[tid * 32 + j] = v;
  }
  __syncthreads();
  if (tid < K_) {                           // T[k] = Σ_win a[k,p]
    float s = 0.f;
#pragma unroll
    for (int j = 0; j < 25; ++j) s += bf2f(sas[tid * 32 + j]);
    Ts[tid] = s;
  }
  __syncthreads();

  v8f acc[8];
#pragma unroll
  for (int nt = 0; nt < 8; ++nt) acc[nt] = v8f{0,0,0,0,0,0,0,0};
  v16bf af = load_a_bf16(sas + (16 * w) * 32, 32, lane);
#pragma unroll
  for (int nt = 0; nt < 8; ++nt) {
    v16bf bf = load_bt_bf16(xct + (16 * nt) * 32, 32, lane);
    acc[nt] = wmma_bf16(af, bf, acc[nt]);
  }
#pragma unroll
  for (int nt = 0; nt < 8; ++nt) store_c_f32(ps + (16 * w) * C_ + 16 * nt, C_, lane, acc[nt]);
  __syncthreads();

#pragma unroll 4
  for (int k = 0; k < K_; ++k) {            // pooled = (B − c·T)/25
    float v = (ps[k * C_ + tid] - cent[k * C_ + tid] * Ts[k]) * 0.04f;
    ps[k * C_ + tid] = v;
  }
  __syncthreads();
  if (tid < K_) {
    float ss = 0.f;
#pragma unroll 8
    for (int c = 0; c < C_; ++c) { float v = ps[tid * C_ + c]; ss += v * v; }
    float r = 1.f / fmaxf(sqrtf(ss), EPS_);
    rr[tid] = r; red2[tid] = ss * r * r;
  }
  __syncthreads();
  if (tid == 0) {
    float g2 = 0.f;
    for (int k = 0; k < K_; ++k) g2 += red2[k];
    gval = 1.f / fmaxf(sqrtf(g2), EPS_);
  }
  __syncthreads();
  float g = gval;
#pragma unroll 4
  for (int k = 0; k < K_; ++k)              // out[n][k*C+c][win]
    out[(((size_t)n * K_ + k) * C_ + tid) * WIN_ + win] = ps[k * C_ + tid] * rr[k] * g;
}

extern "C" void kernel_launch(void* const* d_in, const int* in_sizes, int n_in,
                              void* d_out, int out_size, void* d_ws, size_t ws_size,
                              hipStream_t stream) {
  const float* x      = (const float*)d_in[0];   // [4,128,40,40]
  const float* conv_w = (const float*)d_in[1];   // [64,128]
  const float* cent   = (const float*)d_in[2];   // [64,128]
  float* out = (float*)d_out;
  (void)in_sizes; (void)n_in; (void)out_size; (void)ws_size;

  char* ws = (char*)d_ws;
  unsigned short* xn_pm = (unsigned short*)ws;                 // N*P*C bf16 = 1,638,400 B
  unsigned short* xn_cm = (unsigned short*)(ws + 1638400);     // N*C*P bf16 = 1,638,400 B
  unsigned short* sa    = (unsigned short*)(ws + 3276800);     // N*K*P bf16 =   819,200 B
  unsigned short* wb    = (unsigned short*)(ws + 4096000);     // K*C bf16   =    16,384 B
  float*          S     = (float*)(ws + 4112384);              // N*K f32

  k_prep<<<(K_ * C_ + 255) / 256, 256, 0, stream>>>(conv_w, wb, S);
  k_normalize<<<(N_ * P_) / 256, 256, 0, stream>>>(x, xn_pm, xn_cm);
  k_assign<<<N_ * (P_ / 64), 128, 0, stream>>>(xn_pm, wb, sa, S);
  k_vlad_full<<<N_, 128, 0, stream>>>(sa, xn_cm, cent, S, out + (size_t)N_ * K_ * C_ * WIN_);
  k_windows<<<N_ * WIN_, 128, 0, stream>>>(sa, xn_cm, cent, out);
}
